// STAN_73083163508776
// MI455X (gfx1250) — matrix operand order repaired and checked by reading, not compiled, not run on hardware
//
#include <hip/hip_runtime.h>

typedef __attribute__((ext_vector_type(16))) __bf16 v16bf;
typedef __attribute__((ext_vector_type(8)))  __bf16 v8bf;
typedef __attribute__((ext_vector_type(8)))  float  v8f;
typedef int i32x4 __attribute__((vector_size(16)));   // matches async-LDS builtin param

#define B_   2
#define C_   64
#define T_   7
#define H_   180
#define W_   320
#define R_   16
#define BT_  (B_*T_)          // 14
#define HW_  (H_*W_)          // 57600
#define KB_  18               // 576 / 32
#define NPIX (BT_*HW_)        // 806400
#define NELT ((long)NPIX*C_)  // 51609600

// LDS halo tile for the conv: 3 rows x 36 px x 72 bf16 (64 ch + 8 pad -> 144B
// pixel stride => conflict-free b128 reads across 16 lanes)
#define LPX  72
#define LROW (36*LPX)
#define LTOT (3*LROW)         // 7776 bf16 = 15552 B

#if __has_builtin(__builtin_amdgcn_global_load_async_to_lds_b128)
#define HAVE_ASYNC_LDS 1
#else
#define HAVE_ASYNC_LDS 0
#endif

__device__ __forceinline__ void wait_async0() {
#if __has_builtin(__builtin_amdgcn_s_wait_asynccnt)
  __builtin_amdgcn_s_wait_asynccnt(0);
#else
  asm volatile("s_wait_asynccnt 0x0" ::: "memory");
#endif
}

__device__ __forceinline__ unsigned short f2bf(float f) {
  unsigned u = __builtin_bit_cast(unsigned, f);
  unsigned r = u + 0x7FFFu + ((u >> 16) & 1u);   // round-to-nearest-even
  return (unsigned short)(r >> 16);
}
__device__ __forceinline__ float bf2f(unsigned short h) {
  unsigned u = ((unsigned)h) << 16;
  return __builtin_bit_cast(float, u);
}

// ---------------------------------------------------------------------------
// K0: swizzle tada_w [co][ci][3][3] fp32 -> WMMA A-fragment layout, bf16.
// Abuf[mtile][kb][lane][e];  K order: k = (kh*3+kw)*64 + ci
// A 16x32 bf16 layout: M = lane%16, K(lane,e) = e + 8*((e>=8)+(lane>=16))
// ---------------------------------------------------------------------------
__global__ void k0_weight_prep(const float* __restrict__ w,
                               unsigned short* __restrict__ Abuf) {
  int id = blockIdx.x * blockDim.x + threadIdx.x;   // (mtile*KB_+kb)*32+lane
  if (id >= 4 * KB_ * 32) return;
  int lane = id & 31;
  int kb   = (id >> 5) % KB_;
  int mt   = (id >> 5) / KB_;
  int co   = mt * 16 + (lane & 15);
  unsigned short* dst = Abuf + id * 16;
  for (int e = 0; e < 16; ++e) {
    int kloc = e + 8 * ((e >= 8 ? 1 : 0) + (lane >= 16 ? 1 : 0));
    int kg   = kb * 32 + kloc;
    int tap  = kg >> 6;
    int ci   = kg & 63;
    int kh = tap / 3, kw = tap % 3;
    dst[e] = f2bf(w[((co * C_ + ci) * 3 + kh) * 3 + kw]);
  }
}

// ---------------------------------------------------------------------------
// K1: bilinear warp (zeros padding, pixel coords). Output channels-last bf16.
// ---------------------------------------------------------------------------
__global__ void k1_warp(const float* __restrict__ x, const float* __restrict__ off,
                        unsigned short* __restrict__ warped) {
  int gid = blockIdx.x * blockDim.x + threadIdx.x;
  if (gid >= NPIX) return;
  int f  = gid / HW_;
  int p  = gid - f * HW_;
  int yy = p / W_;
  int xx = p - yy * W_;
  int bi = f / T_, ti = f - bi * T_;
  const float* ofp = off + (long)f * 2 * HW_;
  float pxf = (float)xx + ofp[p];
  float pyf = (float)yy + ofp[HW_ + p];
  float x0 = floorf(pxf), y0 = floorf(pyf);
  float wx1 = pxf - x0, wy1 = pyf - y0;
  float wx0 = 1.f - wx1, wy0 = 1.f - wy1;
  int ix0 = (int)x0, iy0 = (int)y0;
  int ix1 = ix0 + 1, iy1 = iy0 + 1;
  bool vx0 = (ix0 >= 0) && (ix0 < W_), vx1 = (ix1 >= 0) && (ix1 < W_);
  bool vy0 = (iy0 >= 0) && (iy0 < H_), vy1 = (iy1 >= 0) && (iy1 < H_);
  float w00 = wx0 * wy0 * ((vx0 && vy0) ? 1.f : 0.f);
  float w01 = wx1 * wy0 * ((vx1 && vy0) ? 1.f : 0.f);
  float w10 = wx0 * wy1 * ((vx0 && vy1) ? 1.f : 0.f);
  float w11 = wx1 * wy1 * ((vx1 && vy1) ? 1.f : 0.f);
  int cx0 = min(max(ix0, 0), W_ - 1), cx1 = min(max(ix1, 0), W_ - 1);
  int cy0 = min(max(iy0, 0), H_ - 1), cy1 = min(max(iy1, 0), H_ - 1);
  int i00 = cy0 * W_ + cx0, i01 = cy0 * W_ + cx1;
  int i10 = cy1 * W_ + cx0, i11 = cy1 * W_ + cx1;
  int base = (bi * C_ * T_ + ti) * HW_;        // x[b][c][t][h][w], c = 0
  unsigned short* wp = warped + (long)gid * C_;
  for (int c0 = 0; c0 < C_; c0 += 8) {
    unsigned r[4];
    #pragma unroll
    for (int j = 0; j < 4; ++j) {
      const float* xa = x + base + (c0 + 2 * j) * (T_ * HW_);
      const float* xb = xa + (T_ * HW_);
      float va = xa[i00] * w00 + xa[i01] * w01 + xa[i10] * w10 + xa[i11] * w11;
      float vb = xb[i00] * w00 + xb[i01] * w01 + xb[i10] * w10 + xb[i11] * w11;
      r[j] = (unsigned)f2bf(va) | ((unsigned)f2bf(vb) << 16);
    }
    *(uint4*)(wp + c0) = make_uint4(r[0], r[1], r[2], r[3]);
  }
}

// ---------------------------------------------------------------------------
// K2a/K2b: deterministic spatial mean pooling -> apool[b][c][t]
// ---------------------------------------------------------------------------
__global__ void k2a_pool_partial(const unsigned short* __restrict__ warped,
                                 float* __restrict__ partial) {
  int f = blockIdx.y, chunk = blockIdx.x;
  int tid = threadIdx.x;
  int c = tid & 63, g = tid >> 6;
  const unsigned short* wb = warped + (long)f * HW_ * C_;
  float s = 0.f;
  int p0 = chunk * (HW_ / 16) + g;
  for (int i = 0; i < HW_ / 64; ++i) {                 // 900 iters
    int p = p0 + i * 4;
    s += bf2f(wb[(long)p * C_ + c]);
  }
  __shared__ float sm[256];
  sm[tid] = s;
  __syncthreads();
  if (g == 0)
    partial[(f * 16 + chunk) * C_ + c] = sm[c] + sm[64 + c] + sm[128 + c] + sm[192 + c];
}

__global__ void k2b_pool_final(const float* __restrict__ partial,
                               float* __restrict__ apool) {
  int f = blockIdx.x;
  int c = threadIdx.x;                                 // 64 threads
  float s = 0.f;
  for (int ch = 0; ch < 16; ++ch) s += partial[(f * 16 + ch) * C_ + c];
  int bi = f / T_, ti = f - bi * T_;
  apool[(bi * C_ + c) * T_ + ti] = s * (1.f / (float)HW_);
}

// ---------------------------------------------------------------------------
// K2c: full RouteFuncMLP (tiny) -> alpha[b][c][t]
// ---------------------------------------------------------------------------
__global__ void k2c_route(const float* __restrict__ apool,
                          const float* __restrict__ g_w, const float* __restrict__ g_b,
                          const float* __restrict__ a_w, const float* __restrict__ a_b,
                          const float* __restrict__ bn_g, const float* __restrict__ bn_b,
                          const float* __restrict__ bn_rm, const float* __restrict__ bn_rv,
                          const float* __restrict__ b_w, float* __restrict__ alpha) {
  int bi  = blockIdx.x;
  int tid = threadIdx.x;                               // 128 threads
  __shared__ float gp[C_];
  __shared__ float s[C_][T_];
  __shared__ float zz[R_][T_];
  if (tid < C_) {
    float g = 0.f;
    for (int t = 0; t < T_; ++t) g += apool[(bi * C_ + tid) * T_ + t];
    gp[tid] = g * (1.f / (float)T_);
  }
  __syncthreads();
  if (tid < C_) {
    float acc = g_b[tid];
    for (int k = 0; k < C_; ++k) acc += g_w[tid * C_ + k] * gp[k];
    for (int t = 0; t < T_; ++t) s[tid][t] = apool[(bi * C_ + tid) * T_ + t] + acc;
  }
  __syncthreads();
  if (tid < R_ * T_) {
    int r = tid / T_, t = tid - r * T_;
    float z = a_b[r];
    for (int ci = 0; ci < C_; ++ci)
      for (int kk = 0; kk < 3; ++kk) {
        int tt = t + kk - 1;
        if (tt >= 0 && tt < T_) z += a_w[(r * C_ + ci) * 3 + kk] * s[ci][tt];
      }
    float inv = bn_g[r] / sqrtf(bn_rv[r] + 1e-5f);
    z = (z - bn_rm[r]) * inv + bn_b[r];
    zz[r][t] = fmaxf(z, 0.f);
  }
  __syncthreads();
  for (int idx = tid; idx < C_ * T_; idx += blockDim.x) {
    int c = idx / T_, t = idx - c * T_;
    float al = 1.f;
    for (int ri = 0; ri < R_; ++ri)
      for (int kk = 0; kk < 3; ++kk) {
        int tt = t + kk - 1;
        if (tt >= 0 && tt < T_) al += b_w[(c * R_ + ri) * 3 + kk] * zz[ri][tt];
      }
    alpha[(bi * C_ + c) * T_ + t] = al;
  }
}

// ---------------------------------------------------------------------------
// K3: xcal = warped * alpha[b][c][t]  (in place, bf16 channels-last)
// ---------------------------------------------------------------------------
__global__ void k3_scale(unsigned short* __restrict__ xcal,
                         const float* __restrict__ alpha) {
  long gid = (long)blockIdx.x * blockDim.x + threadIdx.x;
  if (gid >= NELT / 4) return;
  long e  = gid * 4;
  int c   = (int)(e & 63);
  long pix = e >> 6;
  int f   = (int)(pix / HW_);
  int bi  = f / T_, ti = f - bi * T_;
  const float* alp = alpha + (bi * C_) * T_ + ti;      // alp[c*T_] = alpha[b][c][t]
  uint2* p = (uint2*)(xcal + e);
  uint2 v = *p;
  unsigned short h0 = (unsigned short)(v.x & 0xFFFF), h1 = (unsigned short)(v.x >> 16);
  unsigned short h2 = (unsigned short)(v.y & 0xFFFF), h3 = (unsigned short)(v.y >> 16);
  h0 = f2bf(bf2f(h0) * alp[(c + 0) * T_]);
  h1 = f2bf(bf2f(h1) * alp[(c + 1) * T_]);
  h2 = f2bf(bf2f(h2) * alp[(c + 2) * T_]);
  h3 = f2bf(bf2f(h3) * alp[(c + 3) * T_]);
  v.x = (unsigned)h0 | ((unsigned)h1 << 16);
  v.y = (unsigned)h2 | ((unsigned)h3 << 16);
  *p = v;
}

// ---------------------------------------------------------------------------
// K4: implicit-GEMM 3x3 conv via v_wmma_f32_16x16x32_bf16.
// Halo (3 x 34 px x 64ch bf16) staged once per block into LDS via the async
// global->LDS path (ASYNCcnt), padded to 144B/pixel for conflict-free ds reads.
// 8 waves: mtile = wid>>1 (c_out/16), ntile = wid&1 (16-pixel strip);
// K loop = 9 taps x 2 c_in halves = 18 WMMAs per wave.
// ---------------------------------------------------------------------------
__global__ __launch_bounds__(256) void k4_conv_wmma(
    const unsigned short* __restrict__ xcal, const unsigned short* __restrict__ Abuf,
    const float* __restrict__ bias, float* __restrict__ out) {
  __shared__ unsigned short ldsB[LTOT];
  int f     = blockIdx.y;
  int strip = blockIdx.x;                      // 0..1799 (10 strips of 32 per row)
  int y  = strip / (W_ / 32);
  int xq = (strip - y * (W_ / 32)) * 32;
  int tid = threadIdx.x;
  int wid = tid >> 5, lane = tid & 31;
  int mtile = wid >> 1, ntile = wid & 1;
  int nlo = lane & 15, khalf = lane >> 4;
  int px = xq + ntile * 16 + nlo;
  const unsigned short* xf = xcal + (long)f * HW_ * C_;
  const unsigned short* ab = Abuf + (mtile * KB_) * 32 * 16;
  __builtin_prefetch(ab + lane * 16, 0, 3);    // global_prefetch_b8

  // zero the halo tile only for border blocks (covers the OOB taps)
  bool border = (y == 0) | (y == H_ - 1) | (xq == 0) | (xq + 32 == W_);
  if (border) {
    int* zp = (int*)ldsB;
    for (int i = tid; i < LTOT / 2; i += 256) zp[i] = 0;
  }
  __syncthreads();

  // stage 3 rows x 34 pixels x 128B in 16B chunks: 816 transfers
  for (int j = 0; j < 4; ++j) {
    int idx = tid + j * 256;
    if (idx < 3 * 34 * 8) {
      int row  = idx / 272;            // 34*8
      int rem  = idx - row * 272;
      int pxl  = rem >> 3;
      int part = rem & 7;
      int sy = y - 1 + row, gx = xq - 1 + pxl;
      if (sy >= 0 && sy < H_ && gx >= 0 && gx < W_) {
        const unsigned short* g = xf + ((long)sy * W_ + gx) * C_ + part * 8;
        unsigned short* l = &ldsB[(row * 36 + pxl) * LPX + part * 8];
#if HAVE_ASYNC_LDS
        __builtin_amdgcn_global_load_async_to_lds_b128(
            (__attribute__((address_space(1))) i32x4*)g,
            (__attribute__((address_space(3))) i32x4*)l, 0, 0);
#else
        *(uint4*)l = *(const uint4*)g;
#endif
      }
    }
  }
#if HAVE_ASYNC_LDS
  wait_async0();
#endif
  __syncthreads();

  v8f acc = {};
  int pbase = ntile * 16 + nlo;                // lds pixel index for dx = -1
  #pragma unroll
  for (int tap = 0; tap < 9; ++tap) {
    int dy = tap / 3, dx = tap % 3;            // lds row / dx offset
    #pragma unroll
    for (int kb2 = 0; kb2 < 2; ++kb2) {
      int kb  = tap * 2 + kb2;
      int cib = kb2 * 32 + khalf * 8;          // c_in cib..+7 and cib+16..+23 per lane
      v16bf a = *(const v16bf*)(ab + (kb * 32 + lane) * 16);
      int offB = (dy * 36 + pbase + dx) * LPX + cib;
      v8bf b0 = *(const v8bf*)&ldsB[offB];
      v8bf b1 = *(const v8bf*)&ldsB[offB + 16];
      v16bf b = __builtin_shufflevector(b0, b1, 0, 1, 2, 3, 4, 5, 6, 7,
                                        8, 9, 10, 11, 12, 13, 14, 15);
      acc = __builtin_amdgcn_wmma_f32_16x16x32_bf16(
          false, a, false, b, (short)0, acc, false, false);
    }
  }

  int bi = f / T_, ti = f - bi * T_;
  #pragma unroll
  for (int r = 0; r < 8; ++r) {
    int co = mtile * 16 + khalf * 8 + r;       // C/D layout: lanes16-31 -> M+8
    float v = acc[r] + bias[co];
    out[((bi * C_ + co) * T_ + ti) * HW_ + y * W_ + px] = v;
  }
}

// ---------------------------------------------------------------------------
extern "C" void kernel_launch(void* const* d_in, const int* in_sizes, int n_in,
                              void* d_out, int out_size, void* d_ws, size_t ws_size,
                              hipStream_t stream) {
  (void)in_sizes; (void)n_in; (void)out_size; (void)ws_size;
  const float* x      = (const float*)d_in[0];
  const float* offset = (const float*)d_in[1];
  const float* g_w    = (const float*)d_in[2];
  const float* g_b    = (const float*)d_in[3];
  const float* a_w    = (const float*)d_in[4];
  const float* a_b    = (const float*)d_in[5];
  const float* bn_g   = (const float*)d_in[6];
  const float* bn_b   = (const float*)d_in[7];
  const float* bn_rm  = (const float*)d_in[8];
  const float* bn_rv  = (const float*)d_in[9];
  const float* b_w    = (const float*)d_in[10];
  const float* tada_w = (const float*)d_in[11];
  const float* tada_b = (const float*)d_in[12];
  float* out = (float*)d_out;

  char* ws = (char*)d_ws;
  unsigned short* xcal = (unsigned short*)ws;                     // 103.2 MB bf16
  size_t o = (size_t)NELT * 2;
  unsigned short* Abuf = (unsigned short*)(ws + o); o += (size_t)4 * KB_ * 32 * 16 * 2;
  float* partial = (float*)(ws + o); o += (size_t)BT_ * 16 * C_ * 4;
  float* apool   = (float*)(ws + o); o += (size_t)B_ * C_ * T_ * 4;
  float* alpha   = (float*)(ws + o); o += (size_t)B_ * C_ * T_ * 4;

  k0_weight_prep<<<9, 256, 0, stream>>>(tada_w, Abuf);
  k1_warp<<<(NPIX + 255) / 256, 256, 0, stream>>>(x, offset, xcal);
  k2a_pool_partial<<<dim3(16, BT_), 256, 0, stream>>>(xcal, partial);
  k2b_pool_final<<<BT_, C_, 0, stream>>>(partial, apool);
  k2c_route<<<B_, 128, 0, stream>>>(apool, g_w, g_b, a_w, a_b,
                                    bn_g, bn_b, bn_rm, bn_rv, b_w, alpha);
  k3_scale<<<(int)((NELT / 4 + 255) / 256), 256, 0, stream>>>(xcal, alpha);
  k4_conv_wmma<<<dim3(HW_ / 32, BT_), 256, 0, stream>>>(xcal, Abuf, tada_b, out);
}
